// WindowAttention_283467841739
// MI455X (gfx1250) — matrix operand order repaired and checked by reading, not compiled, not run on hardware
//
#include <hip/hip_runtime.h>
#include <hip/hip_bf16.h>

typedef __attribute__((ext_vector_type(16))) _Float16 v16h;
typedef __attribute__((ext_vector_type(8)))  _Float16 v8h;
typedef __attribute__((ext_vector_type(8)))  float    v8f;

#define HEADS 8
#define HD    32      // head dim
#define NTOK  64      // tokens per window (8x8)
#define HDIM  128     // H == W
#define CCH   256     // channels
#define SCALE 0.17677669529663687f   // 1/sqrt(32)
#define LOG2E 1.44269504088896340f
#define OSTR  132     // padded row stride (floats) for O staging (bank-conflict free)

// LDS carve (bytes):
//   Qs: [4][64][32] f16 = 16384        (0)
//   Ks: [4][64][32] f16 = 16384        (16384)
//   Vs: [4][32][64] f16 = 16384        (32768)
//   Ps: 4 waves x [64][64] f16 = 32768 (49152)  -- overlaid by Os [64][132] f32 = 33792
//   Bs: [225][4] f32 = 3600            (82944)
#define SMEM_BYTES (82944 + 3600)

__global__ __launch_bounds__(128)
void swin_window_attn_kernel(const float* __restrict__ qg,
                             const float* __restrict__ kg,
                             const float* __restrict__ vg,
                             const float* __restrict__ bt,
                             const int*   __restrict__ shiftp,
                             float* __restrict__ outg)
{
    __shared__ __align__(32) unsigned char smem[SMEM_BYTES];
    _Float16* Qs = (_Float16*)(smem);            // [hl][tok][d]
    _Float16* Ks = (_Float16*)(smem + 16384);    // [hl][tok][d]
    _Float16* Vs = (_Float16*)(smem + 32768);    // [hl][d][tok]
    _Float16* Ps = (_Float16*)(smem + 49152);    // per-wave [n][m]
    float*    Os = (float*)   (smem + 49152);    // overlay: [tok][c_local(128)] stride OSTR
    float*    Bs = (float*)   (smem + 82944);    // [idx][hl]

    const int t    = threadIdx.x;                // 0..127
    const int bid  = blockIdx.x;                 // 0..4095
    const int wid  = bid >> 1;                   // window id 0..2047
    const int half = bid & 1;                    // which 4-head group
    const int b    = wid >> 8;
    const int wi   = wid & 255;
    const int wy   = wi >> 4;
    const int wx   = wi & 15;
    const int shift = shiftp[0];
    const int cbase = half * 128;                // global channel base for this block

    // ---------------- stage K, V (unshifted windows), vectorized ----------
    #pragma unroll
    for (int i = 0; i < 8; ++i) {
        const int seg = i * 128 + t;             // (local channel, window-row)
        const int cl  = seg >> 3;                // 0..127
        const int r   = seg & 7;
        const int hl  = cl >> 5;
        const int d   = cl & 31;
        const int gbase = ((b * CCH + cbase + cl) * HDIM + (wy * 8 + r)) * HDIM + wx * 8;

        const float4 k0 = *(const float4*)(kg + gbase);
        const float4 k1 = *(const float4*)(kg + gbase + 4);
        const int tokb = r * 8;
        _Float16* kd = Ks + hl * (NTOK * HD) + d;         // [tok][d] scatter
        kd[(tokb + 0) * HD] = (_Float16)k0.x;  kd[(tokb + 1) * HD] = (_Float16)k0.y;
        kd[(tokb + 2) * HD] = (_Float16)k0.z;  kd[(tokb + 3) * HD] = (_Float16)k0.w;
        kd[(tokb + 4) * HD] = (_Float16)k1.x;  kd[(tokb + 5) * HD] = (_Float16)k1.y;
        kd[(tokb + 6) * HD] = (_Float16)k1.z;  kd[(tokb + 7) * HD] = (_Float16)k1.w;

        const float4 v0 = *(const float4*)(vg + gbase);
        const float4 v1 = *(const float4*)(vg + gbase + 4);
        v8h vh;
        vh[0] = (_Float16)v0.x; vh[1] = (_Float16)v0.y;
        vh[2] = (_Float16)v0.z; vh[3] = (_Float16)v0.w;
        vh[4] = (_Float16)v1.x; vh[5] = (_Float16)v1.y;
        vh[6] = (_Float16)v1.z; vh[7] = (_Float16)v1.w;
        *(v8h*)(Vs + hl * (HD * NTOK) + d * NTOK + tokb) = vh;  // [d][tok] contiguous
    }

    // ---------------- stage Q (rolled by -shift => read at +shift) --------
    const int col0 = wx * 8 + shift;
    const bool qvec = (col0 + 7 < HDIM) && ((shift & 3) == 0);  // uniform per block
    #pragma unroll
    for (int i = 0; i < 8; ++i) {
        const int seg = i * 128 + t;
        const int cl  = seg >> 3;
        const int r   = seg & 7;
        const int hl  = cl >> 5;
        const int d   = cl & 31;
        const int row = (wy * 8 + r + shift) & (HDIM - 1);
        const int gbase = ((b * CCH + cbase + cl) * HDIM + row) * HDIM;
        _Float16* qd = Qs + hl * (NTOK * HD) + (r * 8) * HD + d;
        if (qvec) {
            const float4 q0 = *(const float4*)(qg + gbase + col0);
            const float4 q1 = *(const float4*)(qg + gbase + col0 + 4);
            qd[0 * HD] = (_Float16)q0.x;  qd[1 * HD] = (_Float16)q0.y;
            qd[2 * HD] = (_Float16)q0.z;  qd[3 * HD] = (_Float16)q0.w;
            qd[4 * HD] = (_Float16)q1.x;  qd[5 * HD] = (_Float16)q1.y;
            qd[6 * HD] = (_Float16)q1.z;  qd[7 * HD] = (_Float16)q1.w;
        } else {
            #pragma unroll
            for (int cc = 0; cc < 8; ++cc) {
                const int col = (col0 + cc) & (HDIM - 1);
                qd[cc * HD] = (_Float16)qg[gbase + col];
            }
        }
    }

    // ---------------- stage bias table (this block's 4 heads) -------------
    for (int idx = t; idx < 225; idx += 128) {
        const float4 bv = *(const float4*)(bt + idx * HEADS + half * 4);
        *(float4*)(Bs + idx * 4) = bv;
    }

    __syncthreads();

    // ---------------- per-wave attention: wave = local head ---------------
    const int wave = t >> 5;        // 0..3
    const int lane = t & 31;
    const int g    = lane >> 4;
    const int ln   = lane & 15;
    const int hl   = wave;

    const _Float16* Qh = Qs + hl * (NTOK * HD);
    const _Float16* Kh = Ks + hl * (NTOK * HD);
    const _Float16* Vh = Vs + hl * (HD * NTOK);
    _Float16*       Pw = Ps + wave * (NTOK * NTOK);

    // A fragments of K^T: A[m_key][d]; lane m, d = g*16..g*16+15 (contiguous)
    v16h a_k[4];
    #pragma unroll
    for (int mt = 0; mt < 4; ++mt) {
        const int m = mt * 16 + ln;
        const v8h lo = *(const v8h*)(Kh + m * HD + g * 16);
        const v8h hi = *(const v8h*)(Kh + m * HD + g * 16 + 8);
        #pragma unroll
        for (int i = 0; i < 8; ++i) { a_k[mt][i] = lo[i]; a_k[mt][8 + i] = hi[i]; }
    }

    // ---- GEMM1: S'[key][query] = K^T * Q ; softmax over keys; P -> LDS ----
    #pragma unroll
    for (int nt = 0; nt < 4; ++nt) {
        const int n  = nt * 16 + ln;
        const int qy = n >> 3, qx = n & 7;
        const v16h b_q = *(const v16h*)(Qh + n * HD + g * 16);

        v8f acc[4];
        #pragma unroll
        for (int mt = 0; mt < 4; ++mt) {
            v8f zero = {0.f, 0.f, 0.f, 0.f, 0.f, 0.f, 0.f, 0.f};
            acc[mt] = __builtin_amdgcn_wmma_f32_16x16x32_f16(
                false, a_k[mt], false, b_q, (short)0, zero, false, false);
        }

        float rmax = -3.0e38f;
        #pragma unroll
        for (int mt = 0; mt < 4; ++mt) {
            #pragma unroll
            for (int r = 0; r < 8; ++r) {
                const int m  = mt * 16 + g * 8 + r;
                const int ky = m >> 3, kx = m & 7;
                const float bv = Bs[((qy - ky + 7) * 15 + (qx - kx + 7)) * 4 + hl];
                const float s = acc[mt][r] * SCALE + bv;
                acc[mt][r] = s;
                rmax = fmaxf(rmax, s);
            }
        }
        rmax = fmaxf(rmax, __shfl_xor(rmax, 16, 32));

        float rsum = 0.f;
        #pragma unroll
        for (int mt = 0; mt < 4; ++mt) {
            #pragma unroll
            for (int r = 0; r < 8; ++r) {
                const float p = exp2f((acc[mt][r] - rmax) * LOG2E);
                acc[mt][r] = p;
                rsum += p;
            }
        }
        rsum += __shfl_xor(rsum, 16, 32);
        const float rinv = 1.0f / rsum;

        #pragma unroll
        for (int mt = 0; mt < 4; ++mt) {
            v8h ph;
            #pragma unroll
            for (int r = 0; r < 8; ++r) ph[r] = (_Float16)(acc[mt][r] * rinv);
            *(v8h*)(Pw + n * NTOK + mt * 16 + g * 8) = ph;
        }
    }

    // ---- GEMM2: O^T[d][query] = V^T * P (accumulators stay in registers) --
    v16h a_v[2][2];
    #pragma unroll
    for (int dt = 0; dt < 2; ++dt) {
        #pragma unroll
        for (int ks = 0; ks < 2; ++ks) {
            const int d = dt * 16 + ln;
            const v8h lo = *(const v8h*)(Vh + d * NTOK + ks * 32 + g * 16);
            const v8h hi = *(const v8h*)(Vh + d * NTOK + ks * 32 + g * 16 + 8);
            #pragma unroll
            for (int i = 0; i < 8; ++i) { a_v[dt][ks][i] = lo[i]; a_v[dt][ks][8 + i] = hi[i]; }
        }
    }

    v8f o[4][2];
    #pragma unroll
    for (int nt = 0; nt < 4; ++nt) {
        const int n = nt * 16 + ln;
        const v16h b0 = *(const v16h*)(Pw + n * NTOK + g * 16);
        const v16h b1 = *(const v16h*)(Pw + n * NTOK + 32 + g * 16);
        #pragma unroll
        for (int dt = 0; dt < 2; ++dt) {
            v8f zero = {0.f, 0.f, 0.f, 0.f, 0.f, 0.f, 0.f, 0.f};
            o[nt][dt] = __builtin_amdgcn_wmma_f32_16x16x32_f16(
                false, a_v[dt][0], false, b0, (short)0, zero, false, false);
            o[nt][dt] = __builtin_amdgcn_wmma_f32_16x16x32_f16(
                false, a_v[dt][1], false, b1, (short)0, o[nt][dt], false, false);
        }
    }

    __syncthreads();   // all waves done reading Ps -> safe to overlay Os

    // stage O to LDS [tok][c_local] (padded stride), contiguous float4 pairs
    #pragma unroll
    for (int nt = 0; nt < 4; ++nt) {
        const int n = nt * 16 + ln;
        #pragma unroll
        for (int dt = 0; dt < 2; ++dt) {
            const int c = hl * 32 + dt * 16 + g * 8;
            float* op = Os + n * OSTR + c;
            *(float4*)(op)     = make_float4(o[nt][dt][0], o[nt][dt][1], o[nt][dt][2], o[nt][dt][3]);
            *(float4*)(op + 4) = make_float4(o[nt][dt][4], o[nt][dt][5], o[nt][dt][6], o[nt][dt][7]);
        }
    }

    __syncthreads();

    // ---------------- cooperative coalesced output stores ------------------
    const bool ovec = (col0 + 7 < HDIM) && ((shift & 3) == 0);  // uniform per block
    #pragma unroll
    for (int i = 0; i < 8; ++i) {
        const int seg = i * 128 + t;
        const int cl  = seg >> 3;
        const int r   = seg & 7;
        const int oy  = (wy * 8 + r + shift) & (HDIM - 1);
        const int obase = ((b * CCH + cbase + cl) * HDIM + oy) * HDIM;
        float vals[8];
        #pragma unroll
        for (int cc = 0; cc < 8; ++cc) vals[cc] = Os[(r * 8 + cc) * OSTR + cl];
        if (ovec) {
            *(float4*)(outg + obase + col0)     = make_float4(vals[0], vals[1], vals[2], vals[3]);
            *(float4*)(outg + obase + col0 + 4) = make_float4(vals[4], vals[5], vals[6], vals[7]);
        } else {
            #pragma unroll
            for (int cc = 0; cc < 8; ++cc)
                outg[obase + ((col0 + cc) & (HDIM - 1))] = vals[cc];
        }
    }
}

extern "C" void kernel_launch(void* const* d_in, const int* in_sizes, int n_in,
                              void* d_out, int out_size, void* d_ws, size_t ws_size,
                              hipStream_t stream) {
    (void)in_sizes; (void)n_in; (void)out_size; (void)d_ws; (void)ws_size;
    const float* q  = (const float*)d_in[0];
    const float* k  = (const float*)d_in[1];
    const float* v  = (const float*)d_in[2];
    const float* bt = (const float*)d_in[3];
    const int*   sh = (const int*)d_in[4];
    float* out = (float*)d_out;
    // 2048 windows x 2 half-head blocks; 128 threads = 4 waves = 4 heads
    swin_window_attn_kernel<<<4096, 128, 0, stream>>>(q, k, v, bt, sh, out);
}
